// SpearmanClassHead_60705067762293
// MI455X (gfx1250) — matrix operand order repaired and checked by reading, not compiled, not run on hardware
//
#include <hip/hip_runtime.h>
#include <hip/hip_bf16.h>

typedef __attribute__((ext_vector_type(16))) _Float16 v16h;
typedef __attribute__((ext_vector_type(8)))  _Float16 v8h;
typedef __attribute__((ext_vector_type(2)))  _Float16 v2h;
typedef __attribute__((ext_vector_type(8)))  float    v8f;

#define ROW_N 1024
#define SORT_THREADS 512

// ---------------------------------------------------------------------------
// Kernel 1: per-row soft_rank (torchsort l2, reg=1), center + normalize,
// emit f16 rank rows to workspace. One block per (row, which-tensor).
// ---------------------------------------------------------------------------
__global__ __launch_bounds__(SORT_THREADS)
void softrank_kernel(const float* __restrict__ za,
                     const float* __restrict__ zb,
                     _Float16* __restrict__ ra,
                     _Float16* __restrict__ rb,
                     int B) {
    __shared__ float skey[ROW_N];   // sorted values (descending)
    __shared__ int   sidx[ROW_N];   // original positions
    __shared__ float ya[ROW_N];     // y = s - w, then isotonic solution
    __shared__ float psum[ROW_N];   // PAVA pool sums, then unsorted ranks
    __shared__ float pcnt[ROW_N];   // PAVA pool counts, then reduction scratch
    __shared__ float sh_mean, sh_rn;

    const int blk   = blockIdx.x;
    const bool isB  = (blk >= B);
    const int row   = isB ? (blk - B) : blk;
    const float* src = (isB ? zb : za) + (size_t)row * ROW_N;
    _Float16*   dst = (isB ? rb : ra) + (size_t)row * ROW_N;

    const int t = threadIdx.x;

    // vectorized load (one b64 per lane) + init indices
    {
        float2 v = ((const float2*)src)[t];
        skey[2 * t]     = v.x; sidx[2 * t]     = 2 * t;
        skey[2 * t + 1] = v.y; sidx[2 * t + 1] = 2 * t + 1;
        // warm L2 for the row the next block streams (speculative; OOB-safe)
        __builtin_prefetch(src + ROW_N + 2 * t, 0, 0);
    }
    __syncthreads();

    // bitonic sort, descending by key (final merge direction: (i&k)==0 -> desc)
    for (unsigned k = 2; k <= ROW_N; k <<= 1) {
        for (unsigned j = k >> 1; j > 0; j >>= 1) {
            unsigned tt = (unsigned)t;
            unsigned i  = ((tt & ~(j - 1u)) << 1) | (tt & (j - 1u));
            unsigned p  = i | j;
            float ki = skey[i], kp = skey[p];
            bool dirDesc = ((i & k) == 0u);
            bool doswap  = dirDesc ? (ki < kp) : (ki > kp);
            if (doswap) {
                skey[i] = kp; skey[p] = ki;
                int tmp = sidx[i]; sidx[i] = sidx[p]; sidx[p] = tmp;
            }
            __syncthreads();
        }
    }

    // y = s - w, with w = [n, n-1, ..., 1]  (reg_strength = 1)
    {
        int i0 = 2 * t, i1 = 2 * t + 1;
        ya[i0] = skey[i0] - (float)(ROW_N - i0);
        ya[i1] = skey[i1] - (float)(ROW_N - i1);
    }
    __syncthreads();

    // PAVA non-increasing isotonic regression (serial; hidden by occupancy)
    if (t == 0) {
        int top = 0;
        for (int i = 0; i < ROW_N; ++i) {
            float s = ya[i], c = 1.0f;
            // merge while current pool mean exceeds previous pool mean
            while (top > 0 && s * pcnt[top - 1] > psum[top - 1] * c) {
                s += psum[top - 1];
                c += pcnt[top - 1];
                --top;
            }
            psum[top] = s; pcnt[top] = c; ++top;
        }
        int pos = 0;
        for (int p = 0; p < top; ++p) {
            float mean = psum[p] / pcnt[p];
            int c = (int)pcnt[p];
            for (int e = 0; e < c; ++e) ya[pos++] = mean;
        }
    }
    __syncthreads();

    // soft rank (sorted order) = s - iso;  unsort into psum[]
    {
        int i0 = 2 * t, i1 = 2 * t + 1;
        psum[sidx[i0]] = skey[i0] - ya[i0];
        psum[sidx[i1]] = skey[i1] - ya[i1];
    }
    __syncthreads();

    // mean reduction (contiguous pairs so the final store can be packed)
    pcnt[t] = psum[2 * t] + psum[2 * t + 1];
    __syncthreads();
    for (int s = SORT_THREADS >> 1; s > 0; s >>= 1) {
        if (t < s) pcnt[t] += pcnt[t + s];
        __syncthreads();
    }
    if (t == 0) sh_mean = pcnt[0] * (1.0f / (float)ROW_N);
    __syncthreads();

    const float m  = sh_mean;
    const float c0 = psum[2 * t]     - m;
    const float c1 = psum[2 * t + 1] - m;

    // norm reduction
    pcnt[t] = c0 * c0 + c1 * c1;
    __syncthreads();
    for (int s = SORT_THREADS >> 1; s > 0; s >>= 1) {
        if (t < s) pcnt[t] += pcnt[t + s];
        __syncthreads();
    }
    if (t == 0) sh_rn = rsqrtf(pcnt[0]);
    __syncthreads();

    const float rn = sh_rn;
    v2h outv;
    outv[0] = (_Float16)(c0 * rn);
    outv[1] = (_Float16)(c1 * rn);
    *(v2h*)(dst + 2 * t) = outv;    // single b32 store per lane
}

// ---------------------------------------------------------------------------
// Kernel 2: batched per-row dot products via v_wmma_f32_16x16x32_f16.
// One wave handles 16 rows: D = A (pred 16xK) * B (target K x 16) accumulated
// over K=1024; the wanted dot products are the diagonal of D.
// A-frag: lane M = lane&15, K-base = (lane>=16)?16:0 (two 16B loads / chunk)
// B-frag: lane N = lane&15, K-off  = (lane>=16)?16:0 (one 32B load / chunk)
// Software-pipelined: next chunk's fragments are loaded before this chunk's
// WMMA so the scheduler can overlap VMEM with the matrix pipe.
// ---------------------------------------------------------------------------
__global__ __launch_bounds__(256)
void corr_wmma_kernel(const _Float16* __restrict__ ra,
                      const _Float16* __restrict__ rb,
                      const float* __restrict__ fc_w,
                      const float* __restrict__ fc_b,
                      float* __restrict__ out,
                      int B) {
    const int lane  = threadIdx.x & 31;
    const int wave  = (blockIdx.x * (blockDim.x >> 5)) + (threadIdx.x >> 5);
    const int row0  = wave * 16;
    if (row0 >= B) return;   // wave-uniform exit (EXEC stays all-ones for WMMA)

    const int mrow   = lane & 15;
    const int kbaseA = (lane >> 4) * 16;   // A: upper half-lanes hold K+16
    const int koffB  = (lane >> 4) * 16;   // B: upper half-lanes hold K+16

    const _Float16* ap = ra + (size_t)(row0 + mrow) * ROW_N + kbaseA;
    const _Float16* bp = rb + (size_t)(row0 + mrow) * ROW_N + koffB;

    // prologue: fragment for chunk 0
    v8h  alo = *(const v8h*)(ap);
    v8h  ahi = *(const v8h*)(ap + 8);
    v16h b   = *(const v16h*)(bp);

    v8f acc = {};
    #pragma unroll 4
    for (int kc = 0; kc < ROW_N - 32; kc += 32) {
        // issue next chunk's loads before this chunk's WMMA
        v8h  nalo = *(const v8h*)(ap + kc + 32);
        v8h  nahi = *(const v8h*)(ap + kc + 40);
        v16h nb   = *(const v16h*)(bp + kc + 32);
        // speculative prefetch (TH=0): silently dropped if OOB
        __builtin_prefetch(ap + kc + 288, 0, 0);
        __builtin_prefetch(bp + kc + 288, 0, 0);

        v16h a;
        #pragma unroll
        for (int i = 0; i < 8; ++i) { a[i] = alo[i]; a[8 + i] = ahi[i]; }
        acc = __builtin_amdgcn_wmma_f32_16x16x32_f16(
                  false, a, false, b, (short)0, acc, false, false);

        alo = nalo; ahi = nahi; b = nb;
    }
    // epilogue: last chunk
    {
        v16h a;
        #pragma unroll
        for (int i = 0; i < 8; ++i) { a[i] = alo[i]; a[8 + i] = ahi[i]; }
        acc = __builtin_amdgcn_wmma_f32_16x16x32_f16(
                  false, a, false, b, (short)0, acc, false, false);
    }

    // Diagonal extraction per C/D layout:
    //   lanes 0..7  : D[m][m] in acc[lane]       (m = lane)
    //   lanes 24..31: D[m][m] in acc[lane-24]    (m = lane-16)
    const bool active = (lane < 8) || (lane >= 24);
    const int  g = (lane < 16) ? lane : (lane - 24);
    float v = acc[0];
    v = (g == 1) ? acc[1] : v;
    v = (g == 2) ? acc[2] : v;
    v = (g == 3) ? acc[3] : v;
    v = (g == 4) ? acc[4] : v;
    v = (g == 5) ? acc[5] : v;
    v = (g == 6) ? acc[6] : v;
    v = (g == 7) ? acc[7] : v;

    if (active) {
        const int m = (lane < 8) ? lane : (lane - 16);
        out[row0 + m] = fabsf(v) * fc_w[0] + fc_b[0];
    }
}

// ---------------------------------------------------------------------------
extern "C" void kernel_launch(void* const* d_in, const int* in_sizes, int n_in,
                              void* d_out, int out_size, void* d_ws, size_t ws_size,
                              hipStream_t stream) {
    const float* z_a  = (const float*)d_in[0];
    const float* z_b  = (const float*)d_in[1];
    const float* fc_w = (const float*)d_in[2];
    const float* fc_b = (const float*)d_in[3];
    float* out = (float*)d_out;

    const int B = in_sizes[0] / ROW_N;   // 8192

    _Float16* ra = (_Float16*)d_ws;
    _Float16* rb = (_Float16*)((char*)d_ws + (size_t)B * ROW_N * sizeof(_Float16));

    // Stage 1: 2*B blocks, one row each (soft_rank + normalize, fp32 -> f16 ws)
    softrank_kernel<<<dim3(2 * B), dim3(SORT_THREADS), 0, stream>>>(z_a, z_b, ra, rb, B);

    // Stage 2: one wave per 16 rows; 8 waves per block
    const int waves  = (B + 15) / 16;
    const int blocks = (waves + 7) / 8;
    corr_wmma_kernel<<<dim3(blocks), dim3(256), 0, stream>>>(ra, rb, fc_w, fc_b, out, B);
}
